// AttentionalGNN_88210038325765
// MI455X (gfx1250) — matrix operand order repaired and checked by reading, not compiled, not run on hardware
//
#include <hip/hip_runtime.h>
#include <hip/hip_bf16.h>
#include <math.h>

typedef __attribute__((ext_vector_type(16))) _Float16 v16h;
typedef __attribute__((ext_vector_type(8)))  _Float16 v8h;
typedef __attribute__((ext_vector_type(8)))  float    v8f;

namespace {
constexpr int Bb   = 4;     // batch
constexpr int Dd   = 128;   // feature dim
constexpr int Nn   = 1024;  // points
constexpr int DALL = 256;   // hamming descriptor dim
constexpr int Hh   = 4;     // heads
constexpr int DIMh = 32;    // per-head dim
constexpr int MIDP = 16;    // D/8 (proj bottleneck)
constexpr int MIDM = 32;    // 2D/8 (mlp bottleneck)
constexpr float EPSBN = 1e-5f;

constexpr int BM = 128, BN = 64, BK = 32;
constexpr int LDP = BK + 8;   // padded LDS row pitch (halves): 80B, 16B-aligned
}

// ---------------------------------------------------------------------------
// Stage one BMxBK A-tile and BNxBK B-tile into LDS (row pitch LDP).
// Path selected at compile time:
//   P==0 : unit-k stride -> 16B vector copies (b128 both sides)
//   P==1 : unit-m/n stride -> m/n-fastest order (coalesced u16 reads)
//   P==2 : generic strides
// K % BK == 0 assumed (true for all GEMMs in this model); M/N guarded by
// uniform fullM/fullN so interior blocks are unpredicated.
// ---------------------------------------------------------------------------
template <int AP, int BP>
__device__ __forceinline__ void stage_tiles(
    const _Float16* __restrict__ A, const _Float16* __restrict__ B,
    _Float16* sA, _Float16* sB,
    int blockM, int blockN, int k0, int M, int Nd,
    long long sAm, long long sAk, long long sBk, long long sBn, int tid)
{
  const bool fullM = (blockM + BM <= M);
  const bool fullN = (blockN + BN <= Nd);

  // ---- A tile -> sA[m*LDP + k] ----
  if (AP == 0) {
#pragma unroll
    for (int i = 0; i < BM * (BK / 8) / 256; ++i) {
      const int e = tid + i * 256;
      const int m = e >> 2, kc = (e & 3) * 8;
      const int mg = blockM + m;
      v8h v;
      if (fullM || mg < M) {
        v = *reinterpret_cast<const v8h*>(A + (long long)mg * sAm + (k0 + kc));
      } else {
#pragma unroll
        for (int j = 0; j < 8; ++j) v[j] = (_Float16)0.f;
      }
      *reinterpret_cast<v8h*>(sA + m * LDP + kc) = v;
    }
  } else if (AP == 1) {
#pragma unroll
    for (int i = 0; i < BM * BK / 256; ++i) {
      const int e = tid + i * 256;
      const int k = e >> 7, m = e & (BM - 1);
      const int mg = blockM + m;
      _Float16 v = (_Float16)0.f;
      if (fullM || mg < M) v = A[(long long)mg + (long long)(k0 + k) * sAk];
      sA[m * LDP + k] = v;
    }
  } else {
#pragma unroll
    for (int i = 0; i < BM * BK / 256; ++i) {
      const int e = tid + i * 256;
      const int m = e >> 5, k = e & 31;
      const int mg = blockM + m;
      _Float16 v = (_Float16)0.f;
      if (fullM || mg < M)
        v = A[(long long)mg * sAm + (long long)(k0 + k) * sAk];
      sA[m * LDP + k] = v;
    }
  }

  // ---- B tile -> sB[n*LDP + k] (transposed stage) ----
  if (BP == 0) {
#pragma unroll
    for (int i = 0; i < BN * (BK / 8) / 256; ++i) {
      const int e = tid + i * 256;
      const int n = e >> 2, kc = (e & 3) * 8;
      const int ng = blockN + n;
      v8h v;
      if (fullN || ng < Nd) {
        v = *reinterpret_cast<const v8h*>(B + (long long)ng * sBn + (k0 + kc));
      } else {
#pragma unroll
        for (int j = 0; j < 8; ++j) v[j] = (_Float16)0.f;
      }
      *reinterpret_cast<v8h*>(sB + n * LDP + kc) = v;
    }
  } else if (BP == 1) {
#pragma unroll
    for (int i = 0; i < BN * BK / 256; ++i) {
      const int e = tid + i * 256;
      const int k = e >> 6, n = e & (BN - 1);
      const int ng = blockN + n;
      _Float16 v = (_Float16)0.f;
      if (fullN || ng < Nd) v = B[(long long)(k0 + k) * sBk + ng];
      sB[n * LDP + k] = v;
    }
  } else {
#pragma unroll
    for (int i = 0; i < BN * BK / 256; ++i) {
      const int e = tid + i * 256;
      const int n = e >> 5, k = e & 31;
      const int ng = blockN + n;
      _Float16 v = (_Float16)0.f;
      if (fullN || ng < Nd)
        v = B[(long long)(k0 + k) * sBk + (long long)ng * sBn];
      sB[n * LDP + k] = v;
    }
  }
}

// ---------------------------------------------------------------------------
// One compute step: each of the 8 waves owns a 32x32 patch ->
// 2 A-frags x 2 B-frags -> 4 WMMAs, fragments gathered as b128 LDS loads.
// ---------------------------------------------------------------------------
__device__ __forceinline__ void compute_step(
    const _Float16* sA, const _Float16* sB,
    int waveM, int waveN, int laneHalf, bool hiLane, v8f acc[4])
{
  // B fragments: lane holds column n; lanes 0-15 carry K=0..15,
  // lanes 16-31 carry K=16..31 (contiguous halves in sB rows)
  v16h bfrag[2];
#pragma unroll
  for (int bt = 0; bt < 2; ++bt) {
    const _Float16* p =
        &sB[(waveN + bt * 16 + laneHalf) * LDP + (hiLane ? 16 : 0)];
    const v8h* pv = reinterpret_cast<const v8h*>(p);
    v8h lo = pv[0], hi = pv[1];
#pragma unroll
    for (int j = 0; j < 8; ++j) { bfrag[bt][j] = lo[j]; bfrag[bt][8 + j] = hi[j]; }
  }

#pragma unroll
  for (int t = 0; t < 2; ++t) {
    // A fragment: lane holds row m; lanes 0-15: K=0..15,
    // lanes 16-31: K=16..31
    v16h afrag;
    {
      const _Float16* p =
          &sA[(waveM + t * 16 + laneHalf) * LDP + (hiLane ? 16 : 0)];
      const v8h* pv = reinterpret_cast<const v8h*>(p);
      v8h lo = pv[0], hi = pv[1];
#pragma unroll
      for (int j = 0; j < 8; ++j) { afrag[j] = lo[j]; afrag[8 + j] = hi[j]; }
    }
#pragma unroll
    for (int bt = 0; bt < 2; ++bt)
      acc[t * 2 + bt] = __builtin_amdgcn_wmma_f32_16x16x32_f16(
          false, afrag, false, bfrag[bt], (short)0, acc[t * 2 + bt],
          false, false);
  }
}

// ---------------------------------------------------------------------------
// Generic strided, two-level-batched WMMA GEMM (f16 in, f32 accumulate),
// double-buffered LDS with software pipelining (one barrier per K-step).
//   C[m,n] = alpha * sum_k A[m,k] * B[k,n]            (mode 0)
//   hamming epilogue with k-split accumulators        (mode 1)
// ksplit uniform, multiple of BK; K-loop statically split over accS/accD.
// ---------------------------------------------------------------------------
template <int AP, int BP>
__global__ __launch_bounds__(256) void wmma_gemm_kernel(
    const _Float16* __restrict__ A, const _Float16* __restrict__ B,
    float* __restrict__ C,
    int M, int Nd, int K,
    long long sAm, long long sAk, long long sBk, long long sBn,
    long long sCm, long long sCn,
    long long bA1, long long bA2, long long bB1, long long bB2,
    long long bC1, long long bC2,
    int batch2, float alpha, int mode, int ksplit)
{
  __shared__ __align__(16) _Float16 lA[2][BM * LDP];
  __shared__ __align__(16) _Float16 lB[2][BN * LDP];

  const int tid = threadIdx.x;
  const int z   = blockIdx.z;
  const int b1  = z / batch2;
  const int b2  = z - b1 * batch2;
  A += b1 * bA1 + b2 * bA2;
  B += b1 * bB1 + b2 * bB2;
  C += b1 * bC1 + b2 * bC2;

  const int blockM = blockIdx.x * BM;
  const int blockN = blockIdx.y * BN;

  const int wave     = tid >> 5;
  const int lane     = tid & 31;
  const int laneHalf = lane & 15;
  const bool hiLane  = lane >= 16;
  const int waveM    = (wave >> 1) * 32;   // 4 waves tile M (each 32 rows)
  const int waveN    = (wave & 1) * 32;    // 2 waves tile N (each 32 cols)

  v8f accS[4], accD[4];
#pragma unroll
  for (int t = 0; t < 4; ++t)
#pragma unroll
    for (int r = 0; r < 8; ++r) { accS[t][r] = 0.f; accD[t][r] = 0.f; }

  const int nsteps    = K / BK;
  const int splitStep = ksplit / BK;

  // prologue: stage step 0
  stage_tiles<AP, BP>(A, B, lA[0], lB[0], blockM, blockN, 0, M, Nd,
                      sAm, sAk, sBk, sBn, tid);
  __syncthreads();

  int s = 0;
  for (; s < splitStep; ++s) {
    const int p = s & 1;
    if (s + 1 < nsteps)
      stage_tiles<AP, BP>(A, B, lA[1 - p], lB[1 - p], blockM, blockN,
                          (s + 1) * BK, M, Nd, sAm, sAk, sBk, sBn, tid);
    compute_step(lA[p], lB[p], waveM, waveN, laneHalf, hiLane, accS);
    __syncthreads();
  }
  for (; s < nsteps; ++s) {
    const int p = s & 1;
    if (s + 1 < nsteps)
      stage_tiles<AP, BP>(A, B, lA[1 - p], lB[1 - p], blockM, blockN,
                          (s + 1) * BK, M, Nd, sAm, sAk, sBk, sBn, tid);
    compute_step(lA[p], lB[p], waveM, waveN, laneHalf, hiLane, accD);
    __syncthreads();
  }

  // epilogue: C/D layout -> VGPR r: lanes 0-15 (M=r, N=lane%16),
  // lanes 16-31 (M=8+r)
  const float khalf = (float)ksplit;
#pragma unroll
  for (int t = 0; t < 2; ++t) {
#pragma unroll
    for (int bt = 0; bt < 2; ++bt) {
      const int nG = blockN + waveN + bt * 16 + laneHalf;
      const int ti = t * 2 + bt;
#pragma unroll
      for (int r = 0; r < 8; ++r) {
        const int mG = blockM + waveM + t * 16 + (hiLane ? 8 + r : r);
        if (mG < M && nG < Nd) {
          float out;
          if (mode == 1) {
            float v1 = (accS[ti][r] + khalf) * 0.5f;
            float v2 = (accD[ti][r] + khalf) * 0.5f;
            v2 = fmaxf(v2, khalf - v2);
            out = (v1 + v2) / (2.0f * khalf);
          } else {
            out = alpha * (accS[ti][r] + accD[ti][r]);
          }
          C[(long long)mG * sCm + (long long)nG * sCn] = out;
        }
      }
    }
  }
}

// ---------------------------------------------------------------------------
// conv1x1 over channels, optional second (concatenated) input, optional f16 copy
// y[b,o,n] = bias[o] + sum_i w[o,i]*x0[b,i,n] (+ sum_j w[o,C0+j]*x1[b,j,n])
// ---------------------------------------------------------------------------
__global__ __launch_bounds__(256) void conv1x1_kernel(
    const float* __restrict__ x0, int C0, const float* __restrict__ x1, int C1,
    const float* __restrict__ w, const float* __restrict__ bias,
    float* __restrict__ y, _Float16* __restrict__ y16, int Nlen)
{
  const int n = blockIdx.x * 256 + threadIdx.x;
  if (n >= Nlen) return;
  const int o = blockIdx.y;
  const int b = blockIdx.z;
  const int Cin = C0 + C1;
  const float* wr = w + (long long)o * Cin;
  float acc = bias[o];
  const float* p0 = x0 + ((long long)b * C0) * Nlen + n;
  for (int i = 0; i < C0; ++i) acc += wr[i] * p0[(long long)i * Nlen];
  if (x1) {
    const float* p1 = x1 + ((long long)b * C1) * Nlen + n;
    for (int i = 0; i < C1; ++i) acc += wr[C0 + i] * p1[(long long)i * Nlen];
  }
  const long long oi = (((long long)b * gridDim.y) + o) * Nlen + n;
  y[oi] = acc;
  if (y16) y16[oi] = (_Float16)acc;
}

// row softmax: f32 in -> f16 out (one block per row)
__global__ __launch_bounds__(256) void row_softmax_f16_kernel(
    const float* __restrict__ in, _Float16* __restrict__ out, int cols)
{
  const long long row = blockIdx.x;
  const float* x = in + row * cols;
  _Float16* y = out + row * cols;
  __shared__ float red[256];
  const int tid = threadIdx.x;

  float m = -1e30f;
  for (int c = tid; c < cols; c += 256) m = fmaxf(m, x[c]);
  red[tid] = m; __syncthreads();
  for (int s = 128; s > 0; s >>= 1) {
    if (tid < s) red[tid] = fmaxf(red[tid], red[tid + s]);
    __syncthreads();
  }
  m = red[0]; __syncthreads();

  float sum = 0.f;
  for (int c = tid; c < cols; c += 256) sum += __expf(x[c] - m);
  red[tid] = sum; __syncthreads();
  for (int s = 128; s > 0; s >>= 1) {
    if (tid < s) red[tid] += red[tid + s];
    __syncthreads();
  }
  const float inv = 1.0f / red[0];
  for (int c = tid; c < cols; c += 256)
    y[c] = (_Float16)(__expf(x[c] - m) * inv);
}

// training-mode BN stats: one block per channel, biased variance
__global__ __launch_bounds__(256) void bn_stats_kernel(
    const float* __restrict__ x, float* __restrict__ mean,
    float* __restrict__ var, int Bc, int Cc, int Nlen)
{
  const int c = blockIdx.x;
  __shared__ float s1[256], s2[256];
  float a = 0.f, q = 0.f;
  const int tot = Bc * Nlen;
  for (int e = threadIdx.x; e < tot; e += 256) {
    const int b = e / Nlen, n = e - b * Nlen;
    const float v = x[(((long long)b * Cc) + c) * Nlen + n];
    a += v; q += v * v;
  }
  s1[threadIdx.x] = a; s2[threadIdx.x] = q; __syncthreads();
  for (int s = 128; s > 0; s >>= 1) {
    if (threadIdx.x < s) { s1[threadIdx.x] += s1[threadIdx.x + s];
                           s2[threadIdx.x] += s2[threadIdx.x + s]; }
    __syncthreads();
  }
  if (threadIdx.x == 0) {
    const float mu = s1[0] / tot;
    mean[c] = mu;
    var[c] = s2[0] / tot - mu * mu;
  }
}

__global__ __launch_bounds__(256) void bn_relu_kernel(
    const float* __restrict__ x, float* __restrict__ y,
    const float* __restrict__ mean, const float* __restrict__ var,
    const float* __restrict__ gamma, const float* __restrict__ beta,
    int Cc, int Nlen)
{
  const long long i = (long long)blockIdx.x * 256 + threadIdx.x;
  const int c = (int)((i / Nlen) % Cc);
  const float v =
      (x[i] - mean[c]) * rsqrtf(var[c] + EPSBN) * gamma[c] + beta[c];
  y[i] = fmaxf(v, 0.f);
}

__global__ __launch_bounds__(256) void add_kernel(
    const float* __restrict__ a, const float* __restrict__ b,
    float* __restrict__ y)
{
  const long long i = (long long)blockIdx.x * 256 + threadIdx.x;
  y[i] = a[i] + b[i];
}

__global__ __launch_bounds__(256) void cvt_f16_kernel(
    const float* __restrict__ x, _Float16* __restrict__ y)
{
  const long long i = (long long)blockIdx.x * 256 + threadIdx.x;
  y[i] = (_Float16)x[i];
}

// ---------------------------------------------------------------------------
extern "C" void kernel_launch(void* const* d_in, const int* in_sizes, int n_in,
                              void* d_out, int out_size, void* d_ws, size_t ws_size,
                              hipStream_t stream)
{
  (void)in_sizes; (void)n_in; (void)out_size; (void)ws_size;

  const float* desc0     = (const float*)d_in[0];
  const float* desc1     = (const float*)d_in[1];
  const float* desc0_all = (const float*)d_in[2];
  const float* desc1_all = (const float*)d_in[3];

  // ---- params (flattened pytree, reference insertion order) ----
  struct PairP { const float *w1, *b1, *w2, *b2; };
  struct LayerP {
    PairP proj[3]; PairP merge;
    const float *mw1, *mb1, *mg, *mbt, *mw2, *mb2;
  };
  LayerP L[6];
  {
    int pi = 4;
    auto nf = [&]() { return (const float*)d_in[pi++]; };
    auto np = [&]() { PairP p; p.w1 = nf(); p.b1 = nf(); p.w2 = nf(); p.b2 = nf(); return p; };
    for (int l = 0; l < 6; ++l) {
      L[l].proj[0] = np(); L[l].proj[1] = np(); L[l].proj[2] = np();
      L[l].merge = np();
      L[l].mw1 = nf(); L[l].mb1 = nf(); L[l].mg = nf();
      L[l].mbt = nf(); L[l].mw2 = nf(); L[l].mb2 = nf();
    }
  }

  // ---- workspace bump allocator ----
  char* wp = (char*)d_ws;
  auto alloc = [&](size_t bytes) {
    void* p = (void*)wp;
    wp += (bytes + 255) & ~(size_t)255;
    return p;
  };
  const long long descN = (long long)Bb * Dd * Nn;        // 524288
  const long long allN  = (long long)Bb * DALL * Nn;      // 1048576

  float*    cur0  = (float*)alloc(descN * 4);
  float*    cur1  = (float*)alloc(descN * 4);
  _Float16* a16_0 = (_Float16*)alloc(allN * 2);
  _Float16* a16_1 = (_Float16*)alloc(allN * 2);
  float*    mid   = (float*)alloc((long long)Bb * MIDP * Nn * 4);
  float*    f32s  = (float*)alloc(descN * 4);             // conv f32 scratch for q/k/v
  _Float16* q16   = (_Float16*)alloc(descN * 2);
  _Float16* k16   = (_Float16*)alloc(descN * 2);
  _Float16* v16b  = (_Float16*)alloc(descN * 2);
  float*    attn  = (float*)alloc(descN * 4);
  float*    mrg   = (float*)alloc(descN * 4);
  float*    hid   = (float*)alloc((long long)Bb * MIDM * Nn * 4);
  float*    dlt0  = (float*)alloc(descN * 4);
  float*    dlt1  = (float*)alloc(descN * 4);
  float*    meanb = (float*)alloc(MIDM * 4);
  float*    varb  = (float*)alloc(MIDM * 4);
  float*    sim   = (float*)alloc((long long)Bb * Hh * Nn * Nn * 4);    // 64 MB
  _Float16* prob  = (_Float16*)alloc((long long)Bb * Hh * Nn * Nn * 2); // 32 MB

  // ---- launch helpers ----
  auto conv = [&](const float* x0, int C0, const float* x1, int C1,
                  const float* w, const float* bias, float* y,
                  _Float16* y16, int Cout) {
    dim3 g(Nn / 256, Cout, Bb);
    conv1x1_kernel<<<g, 256, 0, stream>>>(x0, C0, x1, C1, w, bias, y, y16, Nn);
  };
  auto proj_pair = [&](const float* x, const PairP& p, float* y, _Float16* y16) {
    conv(x, Dd, nullptr, 0, p.w1, p.b1, mid, nullptr, MIDP);
    conv(mid, MIDP, nullptr, 0, p.w2, p.b2, y, y16, Dd);
  };
  auto gemm = [&](const _Float16* A, const _Float16* B, float* C,
                  int M, int Nd, int K,
                  long long sAm, long long sAk, long long sBk, long long sBn,
                  long long sCm, long long sCn,
                  long long bA1, long long bA2, long long bB1, long long bB2,
                  long long bC1, long long bC2,
                  int batch1, int batch2, float alpha, int mode, int ksplit) {
    dim3 g((M + BM - 1) / BM, (Nd + BN - 1) / BN, batch1 * batch2);
    if (sAk == 1 && sBk == 1) {
      wmma_gemm_kernel<0, 0><<<g, 256, 0, stream>>>(A, B, C, M, Nd, K,
          sAm, sAk, sBk, sBn, sCm, sCn, bA1, bA2, bB1, bB2, bC1, bC2,
          batch2, alpha, mode, ksplit);
    } else if (sAm == 1 && sBn == 1) {
      wmma_gemm_kernel<1, 1><<<g, 256, 0, stream>>>(A, B, C, M, Nd, K,
          sAm, sAk, sBk, sBn, sCm, sCn, bA1, bA2, bB1, bB2, bC1, bC2,
          batch2, alpha, mode, ksplit);
    } else {
      wmma_gemm_kernel<2, 2><<<g, 256, 0, stream>>>(A, B, C, M, Nd, K,
          sAm, sAk, sBk, sBn, sCm, sCn, bA1, bA2, bB1, bB2, bC1, bC2,
          batch2, alpha, mode, ksplit);
    }
  };
  auto softmax = [&](const float* in, _Float16* out, long long rows) {
    row_softmax_f16_kernel<<<dim3((unsigned)rows), 256, 0, stream>>>(in, out, Nn);
  };

  // ---- init: working copies + f16 hamming descriptors ----
  hipMemcpyAsync(cur0, desc0, descN * 4, hipMemcpyDeviceToDevice, stream);
  hipMemcpyAsync(cur1, desc1, descN * 4, hipMemcpyDeviceToDevice, stream);
  cvt_f16_kernel<<<dim3((unsigned)(allN / 256)), 256, 0, stream>>>(desc0_all, a16_0);
  cvt_f16_kernel<<<dim3((unsigned)(allN / 256)), 256, 0, stream>>>(desc1_all, a16_1);

  const float qscale = 0.17677669529663687f;  // 1/sqrt(32)

  // ---- 6 layers: self, cross, self, cross, self, cross ----
  bool usedSelf = false, usedCross = false;
  for (int li = 0; li < 6; ++li) {
    const bool cross = (li & 1);
    const bool useHam = !(usedSelf && usedCross);   // true for layers 0,1

    // both streams read the pre-layer cur0/cur1; residual adds are deferred
    for (int s = 0; s < 2; ++s) {
      const float*    x    = s ? cur1 : cur0;
      const float*    src  = s ? (cross ? cur0 : cur1) : (cross ? cur1 : cur0);
      const _Float16* xa   = s ? a16_1 : a16_0;
      const _Float16* sa   = s ? (cross ? a16_0 : a16_1) : (cross ? a16_1 : a16_0);
      float*          dlt  = s ? dlt1 : dlt0;
      const LayerP&   Lp   = L[li];

      // v = proj_pair(source)  (f16 copy for the WMMA apply)
      proj_pair(src, Lp.proj[2], f32s, v16b);

      if (useHam) {
        // sim[b,n,m] = hamming(own_all, source_all); fused split-K WMMA GEMM
        gemm(xa, sa, sim, Nn, Nn, DALL,
             /*A(n,d)=xa[b][d][n]*/ 1, Nn, /*B(d,m)=sa[b][d][m]*/ Nn, 1,
             /*C*/ Nn, 1,
             (long long)DALL * Nn, 0, (long long)DALL * Nn, 0,
             (long long)Nn * Nn, 0,
             Bb, 1, 1.0f, /*mode=hamming*/ 1, /*ksplit*/ DALL / 2);
        softmax(sim, prob, (long long)Bb * Nn);
        // out[b,c,n] = sum_m prob[b,n,m] * v[b,c,m]  (prob shared across heads)
        gemm(prob, v16b, attn, Nn, Dd, Nn,
             Nn, 1, 1, Nn, /*C(n,c)->attn[b][c][n]*/ 1, Nn,
             (long long)Nn * Nn, 0, (long long)Dd * Nn, 0,
             (long long)Dd * Nn, 0,
             Bb, 1, 1.0f, 0, 0);
      } else {
        proj_pair(x,   Lp.proj[0], f32s, q16);
        proj_pair(src, Lp.proj[1], f32s, k16);
        // scores[b,h,n,m] = (1/sqrt(DIM)) sum_dim q[b,dim,h,n]*k[b,dim,h,m]
        // channel index = dim*H + h  ->  k-stride 4N, head offset N
        gemm(q16, k16, sim, Nn, Nn, DIMh,
             1, 4LL * Nn, 4LL * Nn, 1, Nn, 1,
             (long long)Dd * Nn, (long long)Nn,
             (long long)Dd * Nn, (long long)Nn,
             (long long)Hh * Nn * Nn, (long long)Nn * Nn,
             Bb, Hh, qscale, 0, 0);
        softmax(sim, prob, (long long)Bb * Hh * Nn);
        // out[b,dim,h,n] = sum_m prob[b,h,n,m] * v[b,dim,h,m]
        gemm(prob, v16b, attn, Nn, DIMh, Nn,
             Nn, 1, 1, 4LL * Nn, 1, 4LL * Nn,
             (long long)Hh * Nn * Nn, (long long)Nn * Nn,
             (long long)Dd * Nn, (long long)Nn,
             (long long)Dd * Nn, (long long)Nn,
             Bb, Hh, 1.0f, 0, 0);
      }

      // merge proj_pair
      proj_pair(attn, Lp.merge, mrg, nullptr);

      // mlp(concat([x, m])) with training-mode BN + ReLU
      conv(x, Dd, mrg, Dd, Lp.mw1, Lp.mb1, hid, nullptr, MIDM);
      bn_stats_kernel<<<dim3(MIDM), 256, 0, stream>>>(hid, meanb, varb, Bb, MIDM, Nn);
      bn_relu_kernel<<<dim3((unsigned)((long long)Bb * MIDM * Nn / 256)), 256, 0, stream>>>(
          hid, hid, meanb, varb, Lp.mg, Lp.mbt, MIDM, Nn);
      conv(hid, MIDM, nullptr, 0, Lp.mw2, Lp.mb2, dlt, nullptr, Dd);
    }

    // deferred residual updates (after all reads of old cur0/cur1)
    add_kernel<<<dim3((unsigned)(descN / 256)), 256, 0, stream>>>(cur0, dlt0, cur0);
    add_kernel<<<dim3((unsigned)(descN / 256)), 256, 0, stream>>>(cur1, dlt1, cur1);

    if (cross) usedCross = true; else usedSelf = true;
  }

  // ---- outputs: (desc0, desc1) concatenated ----
  float* out = (float*)d_out;
  hipMemcpyAsync(out,         cur0, descN * 4, hipMemcpyDeviceToDevice, stream);
  hipMemcpyAsync(out + descN, cur1, descN * 4, hipMemcpyDeviceToDevice, stream);
}
